// xLSTMBlock_36507222016552
// MI455X (gfx1250) — matrix operand order repaired and checked by reading, not compiled, not run on hardware
//
#include <hip/hip_runtime.h>
#include <math.h>

// ---------------- types ----------------
typedef unsigned short u16;
typedef __attribute__((ext_vector_type(16))) __bf16 v16bf;
typedef __attribute__((ext_vector_type(8)))  __bf16 v8bf;
typedef __attribute__((ext_vector_type(8)))  float  v8f;

static constexpr int Bc = 8;
static constexpr int Lc = 4096;
static constexpr int Dc = 1024;
static constexpr int Hc = 64;   // heads
static constexpr int Mc = Bc * Lc; // 32768 rows

// round-to-nearest-even f32 -> bf16
__device__ __forceinline__ u16 f32_to_bf16(float f) {
    unsigned int u = __float_as_uint(f);
    u += 0x7fffu + ((u >> 16) & 1u);
    return (u16)(u >> 16);
}

// ---------------- f32 -> bf16 weight convert ----------------
__global__ void cvt_bf16_kernel(const float* __restrict__ in, u16* __restrict__ out, int n) {
    int i = blockIdx.x * blockDim.x + threadIdx.x;
    if (i < n) out[i] = f32_to_bf16(in[i]);
}

// ---------------- LayerNorm: [rows, D] f32 -> bf16 ----------------
__global__ void ln_kernel(const float* __restrict__ x, const float* __restrict__ gamma,
                          const float* __restrict__ beta, u16* __restrict__ xn) {
    const int row = blockIdx.x;
    const float* xr = x + (size_t)row * Dc;
    float s = 0.f, s2 = 0.f;
    for (int i = threadIdx.x; i < Dc; i += blockDim.x) {
        float t = xr[i];
        s += t; s2 += t * t;
    }
    #pragma unroll
    for (int o = 16; o > 0; o >>= 1) {
        s  += __shfl_down(s,  o, 32);
        s2 += __shfl_down(s2, o, 32);
    }
    __shared__ float shs[8], shs2[8];
    int wid = threadIdx.x >> 5;
    if ((threadIdx.x & 31) == 0) { shs[wid] = s; shs2[wid] = s2; }
    __syncthreads();
    if (threadIdx.x == 0) {
        float a = 0.f, b = 0.f;
        #pragma unroll
        for (int i = 0; i < 8; ++i) { a += shs[i]; b += shs2[i]; }
        shs[0] = a; shs2[0] = b;
    }
    __syncthreads();
    const float mean = shs[0] * (1.0f / Dc);
    const float var  = shs2[0] * (1.0f / Dc) - mean * mean;
    const float rstd = rsqrtf(var + 1e-5f);
    u16* xo = xn + (size_t)row * Dc;
    for (int i = threadIdx.x; i < Dc; i += blockDim.x) {
        float t = (xr[i] - mean) * rstd * gamma[i] + beta[i];
        xo[i] = f32_to_bf16(t);
    }
}

// ---------------- WMMA GEMM: out[M,N] = (A[M,K] @ W[N,K]^T + bias)*scale + res ----------------
// 64x32 output tile per wave (4 M-subtiles x 2 N-subtiles = 8 WMMAs / K-step).
// Software-pipelined K-loop with AFFINE prefetch index (kk+32) and a peeled
// final step, so fragment addressing stays base-pointer + immediate offset.
// __launch_bounds__(256, 1): allow a large VGPR allocation (~190 live regs for
// accs + double-buffered fragments) — latency is hidden inside the wave by the
// pipeline, not by occupancy, so we must not let the allocator spill to scratch.
//
// A-fragment (16-bit A 16x32): lanes 0-15 hold K 0..15,
// lanes 16-31 hold K 16..31 (contiguous, matching B).
__device__ __forceinline__ v16bf load_a_frag(const u16* __restrict__ p, int k) {
    v8bf lo = *(const v8bf*)(p + k);
    v8bf hi = *(const v8bf*)(p + k + 8);
    v16bf r;
    #pragma unroll
    for (int i = 0; i < 8; ++i) { r[i] = lo[i]; r[i + 8] = hi[i]; }
    return r;
}

// B-fragment (32x16, K x N): lane = column n; lanes 0-15 hold K 0..15,
// lanes 16-31 hold K 16..31 — 16 contiguous bf16 along the W row (= column of W^T).
__device__ __forceinline__ v16bf load_b_frag(const u16* __restrict__ p, int k) {
    return *(const v16bf*)(p + k);
}

#define WMMA_BF16(A_, B_, C_) \
    __builtin_amdgcn_wmma_f32_16x16x32_bf16(false, (A_), false, (B_), (short)0, (C_), false, false)

__global__ void __launch_bounds__(256, 1)
gemm_bf16_wmma(const u16* __restrict__ A, const u16* __restrict__ W,
               const float* __restrict__ bias, const float* __restrict__ res,
               float* __restrict__ out, int M, int N, float scale) {
    constexpr int K = Dc;                      // 1024 for every GEMM in this block
    const int lane = threadIdx.x & 31;
    const int wave = (blockIdx.x * blockDim.x + threadIdx.x) >> 5;
    const int tilesN = N >> 5;
    const int tileM = (wave / tilesN) << 6;   // 64 rows per wave
    const int tileN = (wave % tilesN) << 5;   // 32 cols per wave
    if (tileM >= M) return;

    // per-subtile base pointers; in-loop addressing is base + immediate offset
    const int asub = (lane >> 4) << 4;   // A half-select: +0 or +16 along K
    const int bsub = (lane >> 4) << 4;   // B half-select: +0 or +16 along K
    const u16* pa0 = A + (size_t)(tileM +  0 + (lane & 15)) * K + asub;
    const u16* pa1 = A + (size_t)(tileM + 16 + (lane & 15)) * K + asub;
    const u16* pa2 = A + (size_t)(tileM + 32 + (lane & 15)) * K + asub;
    const u16* pa3 = A + (size_t)(tileM + 48 + (lane & 15)) * K + asub;
    const u16* pb0 = W + (size_t)(tileN +  0 + (lane & 15)) * K + bsub;
    const u16* pb1 = W + (size_t)(tileN + 16 + (lane & 15)) * K + bsub;

    v8f acc00 = {}, acc01 = {}, acc10 = {}, acc11 = {};
    v8f acc20 = {}, acc21 = {}, acc30 = {}, acc31 = {};

    // prologue: fragments for kk = 0
    v16bf a0 = load_a_frag(pa0, 0), a1 = load_a_frag(pa1, 0);
    v16bf a2 = load_a_frag(pa2, 0), a3 = load_a_frag(pa3, 0);
    v16bf b0 = load_b_frag(pb0, 0), b1 = load_b_frag(pb1, 0);

    // main loop: prefetch kk+32 (always valid, affine), compute kk
    for (int kk = 0; kk < K - 32; kk += 32) {
        const int kn = kk + 32;
        v16bf na0 = load_a_frag(pa0, kn), na1 = load_a_frag(pa1, kn);
        v16bf na2 = load_a_frag(pa2, kn), na3 = load_a_frag(pa3, kn);
        v16bf nb0 = load_b_frag(pb0, kn), nb1 = load_b_frag(pb1, kn);

        acc00 = WMMA_BF16(a0, b0, acc00);
        acc01 = WMMA_BF16(a0, b1, acc01);
        acc10 = WMMA_BF16(a1, b0, acc10);
        acc11 = WMMA_BF16(a1, b1, acc11);
        acc20 = WMMA_BF16(a2, b0, acc20);
        acc21 = WMMA_BF16(a2, b1, acc21);
        acc30 = WMMA_BF16(a3, b0, acc30);
        acc31 = WMMA_BF16(a3, b1, acc31);

        a0 = na0; a1 = na1; a2 = na2; a3 = na3; b0 = nb0; b1 = nb1;
    }
    // peeled final K-step
    acc00 = WMMA_BF16(a0, b0, acc00);
    acc01 = WMMA_BF16(a0, b1, acc01);
    acc10 = WMMA_BF16(a1, b0, acc10);
    acc11 = WMMA_BF16(a1, b1, acc11);
    acc20 = WMMA_BF16(a2, b0, acc20);
    acc21 = WMMA_BF16(a2, b1, acc21);
    acc30 = WMMA_BF16(a3, b0, acc30);
    acc31 = WMMA_BF16(a3, b1, acc31);

    // C/D layout: vgpr j, lanes 0-15 -> row j, col lane; lanes 16-31 -> row j+8.
    const int rowh = (lane >> 4) * 8;
    const v8f* accs[4][2] = {{&acc00, &acc01}, {&acc10, &acc11},
                             {&acc20, &acc21}, {&acc30, &acc31}};
    #pragma unroll
    for (int tj = 0; tj < 2; ++tj) {
        const int col = tileN + tj * 16 + (lane & 15);
        const float bcol = bias ? bias[col] : 0.f;
        #pragma unroll
        for (int ti = 0; ti < 4; ++ti) {
            const v8f& acc = *accs[ti][tj];
            #pragma unroll
            for (int j = 0; j < 8; ++j) {
                const int row = tileM + ti * 16 + rowh + j;
                float val = (acc[j] + bcol) * scale;
                if (res) val += res[(size_t)row * N + col];
                out[(size_t)row * N + col] = val;
            }
        }
    }
}

// ---------------- sequential mLSTM scan: one wave32 per (b,h) ----------------
// Lane layout mirrors WMMA C/D: e = lane&15 (column of C / index of n,k,q),
// lane half owns rows d = (lane>>4)*8 .. +7. State fully in registers.
// Software-pipelined: step t+1's loads are issued before step t's state update,
// so the per-step critical path is the short VALU chain, not L2 load latency.
__global__ void mlstm_scan_kernel(const float* __restrict__ q, const float* __restrict__ k,
                                  const float* __restrict__ v, const float* __restrict__ ig,
                                  const float* __restrict__ fg, u16* __restrict__ hs) {
    const int wave = (blockIdx.x * blockDim.x + threadIdx.x) >> 5;
    const int lane = threadIdx.x & 31;
    const int b = wave / Hc, h = wave % Hc;
    const int e = lane & 15;
    const int dbase = (lane >> 4) * 8;

    float C[8];
    #pragma unroll
    for (int j = 0; j < 8; ++j) C[j] = 0.f;
    float ne = 0.f, m = 0.f;

    size_t base = (size_t)b * Lc * Dc + h * 16;
    size_t gb   = (size_t)b * Lc * Hc + h;

    // prologue: load step 0
    float  qe  = q[base + e];
    float  ke  = k[base + e];
    float4 qd0 = *(const float4*)(q + base + dbase);
    float4 qd1 = *(const float4*)(q + base + dbase + 4);
    float4 vd0 = *(const float4*)(v + base + dbase);
    float4 vd1 = *(const float4*)(v + base + dbase + 4);
    float  it  = ig[gb];
    float  ft  = fg[gb];

    for (int t = 0; t < Lc; ++t) {
        // capture current step's operands
        const float  c_qe = qe, c_ke = ke, c_it = it, c_ft = ft;
        const float qd[8] = {qd0.x, qd0.y, qd0.z, qd0.w, qd1.x, qd1.y, qd1.z, qd1.w};
        const float vd[8] = {vd0.x, vd0.y, vd0.z, vd0.w, vd1.x, vd1.y, vd1.z, vd1.w};
        const size_t obase = base;

        // prefetch step t+1 (clamped; independent of the state update below)
        const int tn = (t + 1 < Lc) ? (t + 1) : (Lc - 1);
        const size_t nbase = (size_t)b * Lc * Dc + (size_t)tn * Dc + h * 16;
        const size_t ngb   = (size_t)b * Lc * Hc + (size_t)tn * Hc + h;
        qe  = q[nbase + e];
        ke  = k[nbase + e];
        qd0 = *(const float4*)(q + nbase + dbase);
        qd1 = *(const float4*)(q + nbase + dbase + 4);
        vd0 = *(const float4*)(v + nbase + dbase);
        vd1 = *(const float4*)(v + nbase + dbase + 4);
        it  = ig[ngb];
        ft  = fg[ngb];
        base = nbase; gb = ngb;

        // state update for step t
        const float mnew = fmaxf(c_ft + m, c_it);
        const float fe = __expf(c_ft + m - mnew);
        const float ie = __expf(c_it - mnew);
        m = mnew;

        ne = fmaf(fe, ne, ie * c_ke);
        const float ik = ie * c_ke;

        float cq = 0.f;
        #pragma unroll
        for (int j = 0; j < 8; ++j) {
            C[j] = fmaf(fe, C[j], ik * vd[j]);   // C = f*C + i * v k^T (column e)
            cq   = fmaf(C[j], qd[j], cq);        // partial C^T q over this half's rows
        }
        cq += __shfl_xor(cq, 16, 32);            // combine two row-halves -> Cq[e]

        float dn = ne * c_qe;                    // n replicated in both halves
        #pragma unroll
        for (int o = 1; o < 16; o <<= 1) dn += __shfl_xor(dn, o, 32);

        const float hval = cq / fmaxf(fabsf(dn), 1.0f);
        if (lane < 16) hs[obase + e] = f32_to_bf16(hval);
    }
}

// ---------------- launch ----------------
extern "C" void kernel_launch(void* const* d_in, const int* in_sizes, int n_in,
                              void* d_out, int out_size, void* d_ws, size_t ws_size,
                              hipStream_t stream) {
    const float* x     = (const float*)d_in[0];
    const float* gamma = (const float*)d_in[1];
    const float* beta  = (const float*)d_in[2];
    const float* qw    = (const float*)d_in[3];
    const float* qb    = (const float*)d_in[4];
    const float* kw    = (const float*)d_in[5];
    const float* kb    = (const float*)d_in[6];
    const float* vw    = (const float*)d_in[7];
    const float* vb    = (const float*)d_in[8];
    const float* ow    = (const float*)d_in[9];
    const float* ob    = (const float*)d_in[10];
    const float* iw    = (const float*)d_in[11];
    const float* ib    = (const float*)d_in[12];
    const float* fw    = (const float*)d_in[13];
    const float* fb    = (const float*)d_in[14];

    char* ws = (char*)d_ws;
    size_t off = 0;
    auto carve = [&](size_t bytes) -> void* {
        off = (off + 255) & ~(size_t)255;
        void* p = ws + off;
        off += bytes;
        return p;
    };

    u16*   xn   = (u16*)  carve((size_t)Mc * Dc * sizeof(u16));
    u16*   qwb  = (u16*)  carve((size_t)Dc * Dc * sizeof(u16));
    u16*   kwb  = (u16*)  carve((size_t)Dc * Dc * sizeof(u16));
    u16*   vwb  = (u16*)  carve((size_t)Dc * Dc * sizeof(u16));
    u16*   owb  = (u16*)  carve((size_t)Dc * Dc * sizeof(u16));
    u16*   iwb  = (u16*)  carve((size_t)Hc * Dc * sizeof(u16));
    u16*   fwb  = (u16*)  carve((size_t)Hc * Dc * sizeof(u16));
    float* qf   = (float*)carve((size_t)Mc * Dc * sizeof(float));
    float* kf   = (float*)carve((size_t)Mc * Dc * sizeof(float));
    float* vf   = (float*)carve((size_t)Mc * Dc * sizeof(float));
    float* igf  = (float*)carve((size_t)Mc * Hc * sizeof(float));
    float* fgf  = (float*)carve((size_t)Mc * Hc * sizeof(float));
    u16*   hsb  = (u16*)  carve((size_t)Mc * Dc * sizeof(u16));
    (void)ws_size; (void)in_sizes; (void)n_in; (void)out_size;

    // 1) weights -> bf16
    {
        int n = Dc * Dc, g = (n + 255) / 256;
        cvt_bf16_kernel<<<g, 256, 0, stream>>>(qw, qwb, n);
        cvt_bf16_kernel<<<g, 256, 0, stream>>>(kw, kwb, n);
        cvt_bf16_kernel<<<g, 256, 0, stream>>>(vw, vwb, n);
        cvt_bf16_kernel<<<g, 256, 0, stream>>>(ow, owb, n);
        int ng = Hc * Dc, gg = (ng + 255) / 256;
        cvt_bf16_kernel<<<gg, 256, 0, stream>>>(iw, iwb, ng);
        cvt_bf16_kernel<<<gg, 256, 0, stream>>>(fw, fwb, ng);
    }

    // 2) LayerNorm -> bf16 xn
    ln_kernel<<<Mc, 256, 0, stream>>>(x, gamma, beta, xn);

    // 3) projections (WMMA bf16, f32 accumulate); 64x32 tile per wave
    auto gemm = [&](const u16* A, const u16* W, const float* bias, const float* res,
                    float* out, int M, int N, float scale) {
        int waves = (M / 64) * (N / 32);
        gemm_bf16_wmma<<<waves / 8, 256, 0, stream>>>(A, W, bias, res, out, M, N, scale);
    };
    gemm(xn, qwb, qb, nullptr, qf,  Mc, Dc, 1.0f);
    gemm(xn, kwb, kb, nullptr, kf,  Mc, Dc, 0.25f);   // /sqrt(HD=16)
    gemm(xn, vwb, vb, nullptr, vf,  Mc, Dc, 1.0f);
    gemm(xn, iwb, ib, nullptr, igf, Mc, Hc, 1.0f);
    gemm(xn, fwb, fb, nullptr, fgf, Mc, Hc, 1.0f);

    // 4) sequential scan: B*H = 512 waves, 8 waves/block, software-pipelined loads
    mlstm_scan_kernel<<<(Bc * Hc) / 8, 256, 0, stream>>>(qf, kf, vf, igf, fgf, hsb);

    // 5) output projection + bias + residual -> d_out (f32)
    gemm(hsb, owb, ob, x, (float*)d_out, Mc, Dc, 1.0f);
}
